// Preprocess_Grid_Line2window_8022998909490
// MI455X (gfx1250) — matrix operand order repaired and checked by reading, not compiled, not run on hardware
//
#include <hip/hip_runtime.h>
#include <hip/hip_bf16.h>

typedef __attribute__((ext_vector_type(16))) _Float16 v16h;
typedef __attribute__((ext_vector_type(8)))  _Float16 v8h;
typedef __attribute__((ext_vector_type(8)))  float    v8f;

namespace {
constexpr int   kD  = 128;
constexpr int   kHF = 60, kWF = 80, kNP = kHF * kWF;   // 4800 featmap pixels
constexpr int   kN  = 1024;                            // keypoints per (b)
constexpr int   kL  = 128;                             // epipolar samples
constexpr float kSX = 319.5f, kSY = 239.5f;            // (640-1)/2, (480-1)/2
constexpr float kFX = 39.5f,  kFY = 29.5f;             // (80-1)/2,  (60-1)/2
constexpr float kNEG = -1e9f;

// output offsets (floats), reference tuple order
constexpr int O0 = 0,     O1 = 4096,  O2 = 8192,  O3 = 12288, O4 = 16384,
              O5 = 20480, O6 = 24576, O7 = 26624, O8 = 28672, O9 = 30720;

// workspace offsets (bytes)
constexpr size_t WS_DESC = 0;                         // [img][b][n][d] f16   : 1,048,576 B
constexpr size_t WS_FMT  = 1048576;                   // [map][b][p][d] f16   : 4,915,200 B
constexpr size_t WS_CTR  = WS_FMT + 4915200;          // [dir][b][n][2] f32   : 32,768 B
constexpr size_t WS_CORR = WS_CTR + 32768;            // [dir][b][n][p] f32   : 78,643,200 B
} // namespace

__device__ __forceinline__ float wredSum(float v) {
#pragma unroll
  for (int o = 16; o > 0; o >>= 1) v += __shfl_xor(v, o, 32);
  return v;
}
__device__ __forceinline__ float wredMax(float v) {
#pragma unroll
  for (int o = 16; o > 0; o >>= 1) v = fmaxf(v, __shfl_xor(v, o, 32));
  return v;
}

// Bilinear sample of a 60x80 scalar map at normalized coords, zero padding OOB,
// exactly mirroring the reference gather (per-corner in-bounds * weight).
__device__ __forceinline__ float sampleCorr(const float* __restrict__ cm,
                                            float xn, float yn) {
  float x = (xn + 1.f) * kFX;
  float y = (yn + 1.f) * kFY;
  float x0 = floorf(x), y0 = floorf(y);
  float wx1 = x - x0, wx0 = 1.f - wx1;
  float wy1 = y - y0, wy0 = 1.f - wy1;
  float acc = 0.f;
#pragma unroll
  for (int cy = 0; cy < 2; ++cy) {
#pragma unroll
    for (int cx = 0; cx < 2; ++cx) {
      float xc = x0 + (float)cx, yc = y0 + (float)cy;
      bool  inb = (xc >= 0.f) && (xc <= (float)(kWF - 1)) &&
                  (yc >= 0.f) && (yc <= (float)(kHF - 1));
      int xi = (int)fminf(fmaxf(xc, 0.f), (float)(kWF - 1));
      int yi = (int)fminf(fmaxf(yc, 0.f), (float)(kHF - 1));
      float w = (cx ? wx1 : wx0) * (cy ? wy1 : wy0) * (inb ? 1.f : 0.f);
      acc += w * cm[yi * kWF + xi];
    }
  }
  return acc;
}

// ---------------- Kernel 1: scaled, channel-normalized, transposed f16 maps --
// fmT[map][b][p][d] = (T / max(||xf[b,:,p]||, 1e-12)) * xf[b,d,p]
__global__ void prep_fm_kernel(const float* __restrict__ xf1,
                               const float* __restrict__ xf2,
                               const int* __restrict__ epoch,
                               _Float16* __restrict__ fmT) {
  int wid  = (blockIdx.x * blockDim.x + threadIdx.x) >> 5;
  int lane = threadIdx.x & 31;
  int p   = wid % kNP;
  int b   = (wid / kNP) & 1;
  int map = wid / (2 * kNP);                  // 0 -> xf1, 1 -> xf2
  const float* src = (map == 0) ? xf1 : xf2;
  const float* base = src + (size_t)b * kD * kNP;

  float vals[4]; float ss = 0.f;
#pragma unroll
  for (int j = 0; j < 4; ++j) {
    int d = lane + 32 * j;
    float v = base[d * kNP + p];
    vals[j] = v; ss += v * v;
  }
  ss = wredSum(ss);
  float T = fminf(1.f + (float)epoch[0], 50.f);
  float scale = T / fmaxf(sqrtf(ss), 1e-12f);
  _Float16* dst = fmT + ((size_t)(map * 2 + b) * kNP + p) * kD;
#pragma unroll
  for (int j = 0; j < 4; ++j) dst[lane + 32 * j] = (_Float16)(vals[j] * scale);
}

// ---------------- Kernel 2: bilinear descriptors, L2-normalized, f16 ---------
// Also writes pixel-coordinate outputs coord1 / coord2.
__global__ void prep_desc_kernel(const float* __restrict__ xf1,
                                 const float* __restrict__ xf2,
                                 const float* __restrict__ c1n,
                                 const float* __restrict__ c2n,
                                 _Float16* __restrict__ desc,
                                 float* __restrict__ out) {
  int wid  = (blockIdx.x * blockDim.x + threadIdx.x) >> 5;
  int lane = threadIdx.x & 31;
  int n   = wid % kN;
  int b   = (wid / kN) & 1;
  int img = wid / (2 * kN);                   // 0 -> (xf1, c1), 1 -> (xf2, c2)
  const float* cn  = ((img == 0) ? c1n : c2n) + (size_t)(b * kN + n) * 2;
  const float* base = ((img == 0) ? xf1 : xf2) + (size_t)b * kD * kNP;
  float cx = cn[0], cy = cn[1];

  // bilinear footprint at featuremap resolution (uniform across the wave)
  float x = (cx + 1.f) * kFX, y = (cy + 1.f) * kFY;
  float x0 = floorf(x), y0 = floorf(y);
  float wx1 = x - x0, wx0 = 1.f - wx1;
  float wy1 = y - y0, wy0 = 1.f - wy1;
  int   idx[4]; float w[4];
#pragma unroll
  for (int cyi = 0; cyi < 2; ++cyi) {
#pragma unroll
    for (int cxi = 0; cxi < 2; ++cxi) {
      float xc = x0 + (float)cxi, yc = y0 + (float)cyi;
      bool inb = (xc >= 0.f) && (xc <= (float)(kWF - 1)) &&
                 (yc >= 0.f) && (yc <= (float)(kHF - 1));
      int xi = (int)fminf(fmaxf(xc, 0.f), (float)(kWF - 1));
      int yi = (int)fminf(fmaxf(yc, 0.f), (float)(kHF - 1));
      idx[cyi * 2 + cxi] = yi * kWF + xi;
      w[cyi * 2 + cxi]   = (cxi ? wx1 : wx0) * (cyi ? wy1 : wy0) * (inb ? 1.f : 0.f);
    }
  }

  float vals[4]; float ss = 0.f;
#pragma unroll
  for (int j = 0; j < 4; ++j) {
    int d = lane + 32 * j;
    const float* row = base + (size_t)d * kNP;
    float v = w[0] * row[idx[0]] + w[1] * row[idx[1]] +
              w[2] * row[idx[2]] + w[3] * row[idx[3]];
    vals[j] = v; ss += v * v;
  }
  ss = wredSum(ss);
  float inv = 1.f / fmaxf(sqrtf(ss), 1e-12f);
  _Float16* dst = desc + ((size_t)(img * 2 + b) * kN + n) * kD;
#pragma unroll
  for (int j = 0; j < 4; ++j) dst[lane + 32 * j] = (_Float16)(vals[j] * inv);

  if (lane == 0) {
    int o = (img == 0) ? O0 : O1;
    out[o + (b * kN + n) * 2 + 0] = (cx + 1.f) * kSX;
    out[o + (b * kN + n) * 2 + 1] = (cy + 1.f) * kSY;
  }
}

// ---------------- Kernel 3: WMMA GEMM  corr[dir][b][n][p] ---------------------
// dir0: desc(img0) x fmT(map1);  dir1: desc(img1) x fmT(map0)
// One wave per 16x64 output strip: A fragments (16x128) loaded once and reused
// across 4 N-subtiles -> 16 chained v_wmma_f32_16x16x32_f16 per wave,
// ~13 FLOP/B from L2 instead of ~8 with per-tile waves.
__global__ void gemm_corr_kernel(const _Float16* __restrict__ desc,
                                 const _Float16* __restrict__ fmT,
                                 float* __restrict__ corr) {
  int wid  = (blockIdx.x * blockDim.x + threadIdx.x) >> 5;
  int lane = threadIdx.x & 31;
  int st  = wid % 75;             // N strip of 64 pixels (4800 / 64)
  int mt  = (wid / 75) % 64;      // keypoint tile (1024 / 16)
  int b   = (wid / 4800) & 1;
  int dir = wid / 9600;

  const _Float16* A = desc + ((size_t)(dir * 2 + b) * kN + mt * 16) * kD;
  const _Float16* B = fmT  + ((size_t)((1 - dir) * 2 + b) * kNP + st * 64) * kD;

  int m   = lane & 15;            // A row (M) == B col (N) == C col (N)
  int hi  = lane >> 4;            // half-wave select
  int kbA = hi ? 8 : 0;           // A: K groups {kbA..kbA+7, kbA+16..kbA+23}
  int kbB = hi ? 8 : 0;           // B: K groups {kbB..kbB+7, kbB+16..kbB+23}

  union Frag { v16h v; v8h h[2]; };

  // Load the four A K-chunks once; reuse across all 4 N-subtiles.
  Frag a[4];
#pragma unroll
  for (int k0 = 0; k0 < 4; ++k0) {
    const _Float16* ap = A + m * kD + 32 * k0 + kbA;
    a[k0].h[0] = *(const v8h*)(ap);
    a[k0].h[1] = *(const v8h*)(ap + 16);
  }

  float* outbase = corr + ((size_t)(dir * 2 + b) * kN + mt * 16) * kNP + st * 64;
#pragma unroll
  for (int nt = 0; nt < 4; ++nt) {
    v8f c = {};
#pragma unroll
    for (int k0 = 0; k0 < 4; ++k0) {
      const _Float16* bp = B + (nt * 16 + m) * kD + 32 * k0 + kbB;
      Frag bb;
      bb.h[0] = *(const v8h*)(bp);
      bb.h[1] = *(const v8h*)(bp + 16);
      c = __builtin_amdgcn_wmma_f32_16x16x32_f16(false, a[k0].v, false, bb.v,
                                                 (short)0, c, false, false);
    }
    float* outp = outbase + nt * 16;
#pragma unroll
    for (int v = 0; v < 8; ++v) {
      int row = v + (hi ? 8 : 0);
      outp[(size_t)row * kNP + m] = c[v];
    }
  }
}

// ---------------- Kernel 4: epipolar line search -----------------------------
__global__ void epi_kernel(const float* __restrict__ F1,
                           const float* __restrict__ F2,
                           const float* __restrict__ c1n,
                           const float* __restrict__ c2n,
                           const float* __restrict__ corr,
                           float* __restrict__ center,
                           float* __restrict__ out) {
  int wid  = (blockIdx.x * blockDim.x + threadIdx.x) >> 5;
  int lane = threadIdx.x & 31;
  int n   = wid & (kN - 1);
  int b   = (wid >> 10) & 1;
  int dir = wid >> 11;

  const float* F  = ((dir == 0) ? F1 : F2) + b * 9;
  const float* cn = ((dir == 0) ? c1n : c2n) + (size_t)(b * kN + n) * 2;
  float px = (cn[0] + 1.f) * kSX;
  float py = (cn[1] + 1.f) * kSY;
  float la = F[0] * px + F[1] * py + F[2];
  float lb = F[3] * px + F[4] * py + F[5];
  float lc = F[6] * px + F[7] * py + F[8];

  const float eps = 1e-8f;
  bool  use_x = fabsf(lb) >= fabsf(la);
  bool  degen = (fabsf(la) < eps) && (fabsf(lb) < eps);
  float denB  = (fabsf(lb) < eps) ? eps : lb;
  float denA  = (fabsf(la) < eps) ? eps : la;

  const float* cm = corr + ((size_t)(dir * 2 + b) * kN + n) * kNP;

  float cxn[4], cyn[4], lg[4];
  float anyv = 0.f;
#pragma unroll
  for (int j = 0; j < 4; ++j) {
    int l = lane + 32 * j;
    float t = (float)l * (1.f / (float)(kL - 1));
    float xs_h = t * 639.f;
    float ys_h = -(la * xs_h + lc) / denB;
    float ys_v = t * 479.f;
    float xs_v = -(lb * ys_v + lc) / denA;
    float xs = use_x ? xs_h : xs_v;
    float ys = use_x ? ys_h : ys_v;
    bool valid = (xs >= 0.f) && (xs <= 639.f) && (ys >= 0.f) && (ys <= 479.f) && !degen;
    cxn[j] = xs * (1.f / kSX) - 1.f;
    cyn[j] = ys * (1.f / kSY) - 1.f;
    float cv = sampleCorr(cm, cxn[j], cyn[j]);
    lg[j] = valid ? cv : kNEG;
    anyv  = fmaxf(anyv, valid ? 1.f : 0.f);
  }

  float mx4 = fmaxf(fmaxf(lg[0], lg[1]), fmaxf(lg[2], lg[3]));
  float mm  = wredMax(mx4);
  float es[4]; float s = 0.f, sx = 0.f, sy = 0.f;
#pragma unroll
  for (int j = 0; j < 4; ++j) {
    es[j] = expf(lg[j] - mm);
    s += es[j]; sx += es[j] * cxn[j]; sy += es[j] * cyn[j];
  }
  s  = wredSum(s);
  sx = wredSum(sx);
  sy = wredSum(sy);
  anyv = wredMax(anyv);

  if (lane == 0) {
    float meanx = sx / s, meany = sy / s;
    float ccx = fminf(fmaxf(meanx, -0.875f), 0.875f);
    float ccy = fminf(fmaxf(meany, -0.875f), 0.875f);
    center[((dir * 2 + b) * kN + n) * 2 + 0] = ccx;
    center[((dir * 2 + b) * kN + n) * 2 + 1] = ccy;
    int idx = b * kN + n;
    if (dir == 0) {
      out[O4 + idx * 2 + 0] = (meanx + 1.f) * kSX;   // feat1c_corloc_org (pixels)
      out[O4 + idx * 2 + 1] = (meany + 1.f) * kSY;
      out[O8 + idx] = anyv;                          // valid1
    } else {
      out[O5 + idx * 2 + 0] = meanx;                 // c2_org_n (normalized)
      out[O5 + idx * 2 + 1] = meany;
      out[O9 + idx] = anyv;                          // valid2
    }
  }
}

// ---------------- Kernel 5: window soft-argmax + std --------------------------
__global__ void win_kernel(const float* __restrict__ corr,
                           const float* __restrict__ center,
                           float* __restrict__ out) {
  int wid  = (blockIdx.x * blockDim.x + threadIdx.x) >> 5;
  int lane = threadIdx.x & 31;
  int n   = wid & (kN - 1);
  int b   = (wid >> 10) & 1;
  int dir = wid >> 11;

  const float* cm = corr + ((size_t)(dir * 2 + b) * kN + n) * kNP;
  float ccx = center[((dir * 2 + b) * kN + n) * 2 + 0];
  float ccy = center[((dir * 2 + b) * kN + n) * 2 + 1];

  const float step = 0.25f / 15.f;
  float sxn[8], syn[8], lg[8];
#pragma unroll
  for (int j = 0; j < 8; ++j) {
    int k  = lane + 32 * j;
    int ix = k & 15, iy = k >> 4;
    sxn[j] = ccx + (-0.125f + (float)ix * step);
    syn[j] = ccy + (-0.125f + (float)iy * step);
    lg[j]  = sampleCorr(cm, sxn[j], syn[j]);
  }

  float mloc = lg[0];
#pragma unroll
  for (int j = 1; j < 8; ++j) mloc = fmaxf(mloc, lg[j]);
  float mm = wredMax(mloc);

  float s = 0.f, sx = 0.f, sy = 0.f, sx2 = 0.f, sy2 = 0.f;
#pragma unroll
  for (int j = 0; j < 8; ++j) {
    float e = expf(lg[j] - mm);
    s += e; sx += e * sxn[j]; sy += e * syn[j];
    sx2 += e * sxn[j] * sxn[j]; sy2 += e * syn[j] * syn[j];
  }
  s   = wredSum(s);
  sx  = wredSum(sx);  sy  = wredSum(sy);
  sx2 = wredSum(sx2); sy2 = wredSum(sy2);

  if (lane == 0) {
    float mx = sx / s, my = sy / s;
    float vx = sx2 / s - mx * mx;
    float vy = sy2 / s - my * my;
    float std = sqrtf(fmaxf(vx, 0.f) + fmaxf(vy, 0.f) + 1e-12f);
    int idx = b * kN + n;
    if (dir == 0) {
      out[O2 + idx * 2 + 0] = (mx + 1.f) * kSX;      // feat1w_corloc
      out[O2 + idx * 2 + 1] = (my + 1.f) * kSY;
      out[O6 + idx] = std;                            // f1w_std
    } else {
      out[O3 + idx * 2 + 0] = (mx + 1.f) * kSX;      // feat2w_corloc
      out[O3 + idx * 2 + 1] = (my + 1.f) * kSY;
      out[O7 + idx] = std;                            // f2w_std
    }
  }
}

extern "C" void kernel_launch(void* const* d_in, const int* in_sizes, int n_in,
                              void* d_out, int out_size, void* d_ws, size_t ws_size,
                              hipStream_t stream) {
  const float* xf1 = (const float*)d_in[0];
  const float* xf2 = (const float*)d_in[1];
  const float* F1  = (const float*)d_in[2];
  const float* F2  = (const float*)d_in[3];
  const float* c1n = (const float*)d_in[4];
  const float* c2n = (const float*)d_in[5];
  const int*   ep  = (const int*)d_in[6];

  char* ws = (char*)d_ws;
  _Float16* desc = (_Float16*)(ws + WS_DESC);
  _Float16* fmT  = (_Float16*)(ws + WS_FMT);
  float*    ctr  = (float*)(ws + WS_CTR);
  float*    corr = (float*)(ws + WS_CORR);
  float*    out  = (float*)d_out;

  // 1) scaled/normalized transposed maps: 2*2*4800 waves, 8 waves/block
  prep_fm_kernel<<<2400, 256, 0, stream>>>(xf1, xf2, ep, fmT);
  // 2) descriptors + coord outputs: 2*2*1024 waves
  prep_desc_kernel<<<512, 256, 0, stream>>>(xf1, xf2, c1n, c2n, desc, out);
  // 3) WMMA GEMM: 2*2*64*75 = 19200 strip-waves, 16 WMMAs each
  gemm_corr_kernel<<<2400, 256, 0, stream>>>(desc, fmT, corr);
  // 4) epipolar soft-argmax: 2*2*1024 waves
  epi_kernel<<<512, 256, 0, stream>>>(F1, F2, c1n, c2n, corr, ctr, out);
  // 5) window soft-argmax + std: 2*2*1024 waves
  win_kernel<<<512, 256, 0, stream>>>(corr, ctr, out);
}
